// LatentLowRankRNN_30073361006692
// MI455X (gfx1250) — compile-verified, matrix operand order checked by reading, not run on hardware
//
#include <hip/hip_runtime.h>
#include <hip/hip_bf16.h>

#define TT 100
#define DD 128
#define BB 2048
#define ALPHA_C 0.2f
#define NOISE_C 0.05f

typedef __attribute__((ext_vector_type(16))) __bf16 v16bf;
typedef __attribute__((ext_vector_type(8)))  float v8f;
typedef __attribute__((ext_vector_type(4)))  unsigned int u32x4;
typedef __attribute__((ext_vector_type(8)))  int i32x8;
typedef __attribute__((ext_vector_type(4)))  int i32x4;

static __device__ __forceinline__ unsigned short f2bf(float f) {
    unsigned int u = __float_as_uint(f);
    u += 0x7FFFu + ((u >> 16) & 1u);          // round-to-nearest-even
    return (unsigned short)(u >> 16);
}

union Frag { v16bf v; unsigned int u[8]; };

// LDS row pitch in ushorts: 32 data + 8 pad -> 80 bytes (16B-aligned rows,
// row stride = 20 dwords -> conflict-free fragment reads across 16 rows).
// TDM reproduces this pitch: 64B data rows + 16B pad via pad_interval/amount.
#define LDP 40

static __device__ __forceinline__ unsigned lds_off32(const void* p) {
    // Low 32 bits of a flat LDS-aperture address are the LDS byte address.
    return (unsigned)(unsigned long long)p;
}

// Issue one TDM descriptor: 128 rows x 32 bf16 tile (row stride = K elements)
// global -> LDS at lds_byte_addr with 80B destination pitch. Tracked by TENSORcnt.
static __device__ __forceinline__ void tdm_stage_tile(unsigned lds_byte_addr,
                                                      const unsigned short* gaddr,
                                                      int K, int rows)
{
    const unsigned long long ga = (unsigned long long)gaddr;
    u32x4 g0;
    g0[0] = 1u;                                        // count=1, user mode, no gather
    g0[1] = lds_byte_addr;                             // bits 63:32  lds_addr
    g0[2] = (unsigned)(ga & 0xFFFFFFFFu);              // bits 95:64  global_addr[31:0]
    g0[3] = (unsigned)((ga >> 32) & 0x01FFFFFFu)       // bits 120:96 global_addr[56:32]
          | (2u << 30);                                // bits 127:126 type=2 (image)

    i32x8 g1;
    g1[0] = (int)((1u << 16)                           // data_size = 1 -> 2 bytes
                | (1u << 20)                           // pad_enable
                | (3u << 22)                           // pad_interval: 16 dwords (64B)
                | (3u << 25));                         // pad_amount:   4 dwords (16B)
    const unsigned td0 = (unsigned)K;                  // tensor_dim0 (row length, elems)
    const unsigned td1 = (unsigned)rows;               // tensor_dim1 (#rows)
    g1[1] = (int)((td0 & 0xFFFFu) << 16);              // 63:48  tensor_dim0[15:0]
    g1[2] = (int)((td0 >> 16) | ((td1 & 0xFFFFu) << 16)); // 79:64 dim0 hi, 95:80 dim1 lo
    g1[3] = (int)((td1 >> 16) | (32u << 16));          // 111:96 dim1 hi, 127:112 tile_dim0=32
    g1[4] = (int)128;                                  // 143:128 tile_dim1=128, tile_dim2=0
    g1[5] = (int)td0;                                  // 191:160 tensor_dim0_stride[31:0]=K
    g1[6] = 0;                                         // stride hi, tensor_dim1_stride lo
    g1[7] = 0;

    const i32x4 z4 = {0, 0, 0, 0};                     // groups 2/3: 2-D tensor
    const i32x8 z8 = {0, 0, 0, 0, 0, 0, 0, 0};
    __builtin_amdgcn_tensor_load_to_lds(g0, g1, z4, z4, z8, 0);
}

// Load one 16x32 bf16 fragment (A-style layout: lane%16 = row, lane/16 = K-half;
// dword j<4 -> K pair (4*khalf+j), j>=4 -> K pair (8+4*khalf+(j-4)) of the 32-chunk).
static __device__ __forceinline__ void load_frag(const unsigned short* rowbase, int khalf, Frag& f) {
    const unsigned int* p = (const unsigned int*)rowbase;
    const int o = khalf * 4;
#pragma unroll
    for (int j = 0; j < 4; ++j) f.u[j]     = p[o + j];
#pragma unroll
    for (int j = 0; j < 4; ++j) f.u[4 + j] = p[8 + o + j];
}

// C = act(A[MxK](bf16) * Wt[NxK](bf16, pre-transposed) + bias).
// 128x128 tile per workgroup, 8 waves each owning 32x64 (2x4 WMMA tiles),
// K-chunks of 32, LDS double-buffered; staging done by the Tensor Data Mover
// (one descriptor per tile, issued by wave 0, pipelined on TENSORcnt).
template<bool RELU, bool WB16, bool WF32>
__global__ __launch_bounds__(256) void gemm_bf16_wmma(
    const unsigned short* __restrict__ A,
    const unsigned short* __restrict__ Wt,
    const float* __restrict__ bias,
    unsigned short* __restrict__ outB,
    float* __restrict__ outF,
    int M, int N, int K, long long strideF)
{
    __shared__ unsigned short As[2][128 * LDP];
    __shared__ unsigned short Bs[2][128 * LDP];

    const int tid  = threadIdx.x;
    const int lane = tid & 31;
    const int wave = tid >> 5;
    const int bm   = blockIdx.y * 128;
    const int bn   = blockIdx.x * 128;
    const int wm   = (wave >> 1) * 32;
    const int wn   = (wave & 1) * 64;
    const int frow  = lane & 15;
    const int khalf = lane >> 4;

    const unsigned ldsA[2] = { lds_off32(&As[0][0]), lds_off32(&As[1][0]) };
    const unsigned ldsB[2] = { lds_off32(&Bs[0][0]), lds_off32(&Bs[1][0]) };
    const unsigned short* aBase = A  + (size_t)bm * K;
    const unsigned short* bBase = Wt + (size_t)bn * K;

    v8f acc[2][4];
#pragma unroll
    for (int mt = 0; mt < 2; ++mt)
#pragma unroll
        for (int nt = 0; nt < 4; ++nt) { v8f z = {}; acc[mt][nt] = z; }

    const int nk = K >> 5;
    if (wave == 0) {                       // prologue: chunk 0 -> buffer 0
        tdm_stage_tile(ldsA[0], aBase, K, 128);
        tdm_stage_tile(ldsB[0], bBase, K, 128);
    }

    for (int i = 0; i < nk; ++i) {
        const int cur = i & 1;
        if (wave == 0) {
            if (i + 1 < nk) {
                // Stage chunk i+1 into the other buffer (last read of it ended at
                // the trailing barrier of iteration i-1). TDM ops complete in order
                // per wave: wait TENSORcnt<=2 retires chunk i's two descriptors.
                tdm_stage_tile(ldsA[cur ^ 1], aBase + (i + 1) * 32, K, 128);
                tdm_stage_tile(ldsB[cur ^ 1], bBase + (i + 1) * 32, K, 128);
                __builtin_amdgcn_s_wait_tensorcnt(2);
            } else {
                __builtin_amdgcn_s_wait_tensorcnt(0);
            }
        }
        __syncthreads();

        Frag af[2], bfr[4];
#pragma unroll
        for (int mt = 0; mt < 2; ++mt)
            load_frag(&As[cur][(wm + mt * 16 + frow) * LDP], khalf, af[mt]);
#pragma unroll
        for (int nt = 0; nt < 4; ++nt)
            load_frag(&Bs[cur][(wn + nt * 16 + frow) * LDP], khalf, bfr[nt]);

#pragma unroll
        for (int mt = 0; mt < 2; ++mt)
#pragma unroll
            for (int nt = 0; nt < 4; ++nt)
                acc[mt][nt] = __builtin_amdgcn_wmma_f32_16x16x32_bf16(
                    false, af[mt].v, false, bfr[nt].v,
                    (short)0, acc[mt][nt], false, false);
        __syncthreads();
    }

    // Epilogue: bias + optional ReLU; C layout: VGPR i -> M = i + 8*(lane/16), N = lane%16.
    const int nlane = lane & 15;
    const int mhi   = (lane >> 4) << 3;
#pragma unroll
    for (int nt = 0; nt < 4; ++nt) {
        const int n_g = bn + wn + nt * 16 + nlane;
        const float bv = bias[n_g];
#pragma unroll
        for (int mt = 0; mt < 2; ++mt) {
#pragma unroll
            for (int i = 0; i < 8; ++i) {
                const int m_g = bm + wm + mt * 16 + mhi + i;
                float v = acc[mt][nt][i] + bv;
                if (RELU) v = fmaxf(v, 0.0f);
                if (WB16) outB[(size_t)m_g * N + n_g] = f2bf(v);
                if (WF32) outF[(size_t)m_g * strideF + n_g] = v;
            }
        }
    }
}

// One-time weight prep: Wt[n*K + k] = bf16(W[k*N + n]) (transpose + convert).
__global__ void wt_conv_kernel(const float* __restrict__ W, unsigned short* __restrict__ Wt,
                               int K, int N)
{
    const int idx = blockIdx.x * blockDim.x + threadIdx.x;
    if (idx >= K * N) return;
    const int k = idx / N, n = idx - k * N;
    Wt[(size_t)n * K + k] = f2bf(W[idx]);
}

// x -> nan_to_num -> bf16 activations; also x_pred[:,0,:] = cleaned x[:,0,:].
__global__ void prep_kernel(const float* __restrict__ x, unsigned short* __restrict__ Xbf,
                            float* __restrict__ x_pred, long long total)
{
    long long idx = (long long)blockIdx.x * blockDim.x + threadIdx.x;
    if (idx >= total) return;
    float v = x[idx];
    v = (v != v) ? 0.0f : v;
    Xbf[idx] = f2bf(v);
    if (((idx / DD) % TT) == 0) x_pred[idx] = v;
}

// h = h_enc[:,0,:]; z_pred[:,0,:] = h
__global__ void init_rnn_kernel(const float* __restrict__ Henc, float* __restrict__ h,
                                float* __restrict__ z_pred)
{
    const int idx = blockIdx.x * blockDim.x + threadIdx.x;   // BB*DD
    const int b = idx >> 7, j = idx & 127;
    const float v = Henc[(size_t)b * TT * DD + j];
    h[idx] = v;
    z_pred[(size_t)b * TT * DD + j] = v;
}

// One RNN step: rank-2 recurrence + noise; one wave per row, shfl_xor reduction.
__global__ __launch_bounds__(256) void rnn_step_kernel(
    float* __restrict__ h, const float* __restrict__ noise, int s,
    const float* __restrict__ nmat, const float* __restrict__ mmat,
    float* __restrict__ z_pred, unsigned short* __restrict__ Vbf)
{
    const int wave = threadIdx.x >> 5, lane = threadIdx.x & 31;
    const int b = blockIdx.x * 8 + wave;
    float hv[4], t0 = 0.0f, t1 = 0.0f;
#pragma unroll
    for (int q = 0; q < 4; ++q) {
        const int j = q * 32 + lane;
        const float x = h[(size_t)b * DD + j];
        hv[q] = x;
        const float r = tanhf(x);
        t0 += r * nmat[2 * j];
        t1 += r * nmat[2 * j + 1];
    }
#pragma unroll
    for (int o = 16; o >= 1; o >>= 1) {
        t0 += __shfl_xor(t0, o, 32);
        t1 += __shfl_xor(t1, o, 32);
    }
#pragma unroll
    for (int q = 0; q < 4; ++q) {
        const int j = q * 32 + lane;
        const float u  = (t0 * mmat[2 * j] + t1 * mmat[2 * j + 1]) * (1.0f / 128.0f);
        const float nz = noise[((size_t)b * (TT - 1) + s) * DD + j];
        const float hn = hv[q] + ALPHA_C * (u - hv[q]) + NOISE_C * nz;
        h[(size_t)b * DD + j] = hn;
        z_pred[((size_t)b * TT + (s + 1)) * DD + j] = hn;
        Vbf[(size_t)b * DD + j] = f2bf(tanhf(hn));
    }
}

extern "C" void kernel_launch(void* const* d_in, const int* in_sizes, int n_in,
                              void* d_out, int out_size, void* d_ws, size_t ws_size,
                              hipStream_t stream)
{
    const float* x     = (const float*)d_in[0];
    const float* noise = (const float*)d_in[1];
    const float* e_w0  = (const float*)d_in[3];  const float* e_b0 = (const float*)d_in[4];
    const float* e_w1  = (const float*)d_in[5];  const float* e_b1 = (const float*)d_in[6];
    const float* e_w2  = (const float*)d_in[7];  const float* e_b2 = (const float*)d_in[8];
    const float* d_w0  = (const float*)d_in[9];  const float* d_b0 = (const float*)d_in[10];
    const float* d_w1  = (const float*)d_in[11]; const float* d_b1 = (const float*)d_in[12];
    const float* d_w2  = (const float*)d_in[13]; const float* d_b2 = (const float*)d_in[14];
    const float* mmat  = (const float*)d_in[15];
    const float* nmat  = (const float*)d_in[16];

    float* x_pred = (float*)d_out;
    float* x_rec  = x_pred + (size_t)BB * TT * DD;
    float* z_pred = x_rec  + (size_t)BB * TT * DD;

    char* ws = (char*)d_ws; size_t off = 0;
    auto carve = [&](size_t bytes) -> char* {
        char* p = ws + off; off = (off + bytes + 255) & ~(size_t)255; return p;
    };
    const int MT = BB * TT;   // 204800 tokens
    unsigned short* Xbf    = (unsigned short*)carve((size_t)MT * DD  * 2);
    unsigned short* buf512 = (unsigned short*)carve((size_t)MT * 512 * 2);
    unsigned short* buf256 = (unsigned short*)carve((size_t)MT * 256 * 2);
    float*          Henc   = (float*)         carve((size_t)MT * DD  * 4);
    unsigned short* Hencbf = (unsigned short*)carve((size_t)MT * DD  * 2);
    float*          h      = (float*)         carve((size_t)BB * DD  * 4);
    unsigned short* Vbf    = (unsigned short*)carve((size_t)BB * DD  * 2);
    unsigned short* rb256  = (unsigned short*)carve((size_t)BB * 256 * 2);
    unsigned short* rb512  = (unsigned short*)carve((size_t)BB * 512 * 2);
    unsigned short* e0t    = (unsigned short*)carve((size_t)128 * 512 * 2);
    unsigned short* e1t    = (unsigned short*)carve((size_t)512 * 256 * 2);
    unsigned short* e2t    = (unsigned short*)carve((size_t)256 * 128 * 2);
    unsigned short* d0t    = (unsigned short*)carve((size_t)128 * 256 * 2);
    unsigned short* d1t    = (unsigned short*)carve((size_t)256 * 512 * 2);
    unsigned short* d2t    = (unsigned short*)carve((size_t)512 * 128 * 2);

    // One-time weight transpose+convert to bf16 (weights stay hot in L2 after this).
    wt_conv_kernel<<<(128 * 512 + 255) / 256, 256, 0, stream>>>(e_w0, e0t, 128, 512);
    wt_conv_kernel<<<(512 * 256 + 255) / 256, 256, 0, stream>>>(e_w1, e1t, 512, 256);
    wt_conv_kernel<<<(256 * 128 + 255) / 256, 256, 0, stream>>>(e_w2, e2t, 256, 128);
    wt_conv_kernel<<<(128 * 256 + 255) / 256, 256, 0, stream>>>(d_w0, d0t, 128, 256);
    wt_conv_kernel<<<(256 * 512 + 255) / 256, 256, 0, stream>>>(d_w1, d1t, 256, 512);
    wt_conv_kernel<<<(512 * 128 + 255) / 256, 256, 0, stream>>>(d_w2, d2t, 512, 128);

    const long long total = (long long)MT * DD;
    prep_kernel<<<(unsigned)((total + 255) / 256), 256, 0, stream>>>(x, Xbf, x_pred, total);

    const dim3 blk(256);
    // Encoder over all tokens
    gemm_bf16_wmma<true,  true,  false><<<dim3(4, MT / 128), blk, 0, stream>>>(Xbf,    e0t, e_b0, buf512, nullptr, MT, 512, 128, 0);
    gemm_bf16_wmma<true,  true,  false><<<dim3(2, MT / 128), blk, 0, stream>>>(buf512, e1t, e_b1, buf256, nullptr, MT, 256, 512, 0);
    gemm_bf16_wmma<false, true,  true ><<<dim3(1, MT / 128), blk, 0, stream>>>(buf256, e2t, e_b2, Hencbf, Henc,    MT, 128, 256, 128);
    // Decoder -> x_reconstruct
    gemm_bf16_wmma<true,  true,  false><<<dim3(2, MT / 128), blk, 0, stream>>>(Hencbf, d0t, d_b0, buf256, nullptr, MT, 256, 128, 0);
    gemm_bf16_wmma<true,  true,  false><<<dim3(4, MT / 128), blk, 0, stream>>>(buf256, d1t, d_b1, buf512, nullptr, MT, 512, 256, 0);
    gemm_bf16_wmma<false, false, true ><<<dim3(1, MT / 128), blk, 0, stream>>>(buf512, d2t, d_b2, nullptr, x_rec,  MT, 128, 512, 128);

    init_rnn_kernel<<<(BB * DD) / 256, 256, 0, stream>>>(Henc, h, z_pred);

    // Sequential low-rank RNN: 99 dependent steps, decoder per step (M=2048)
    for (int s = 0; s < TT - 1; ++s) {
        rnn_step_kernel<<<BB / 8, 256, 0, stream>>>(h, noise, s, nmat, mmat, z_pred, Vbf);
        gemm_bf16_wmma<true,  true,  false><<<dim3(2, BB / 128), blk, 0, stream>>>(Vbf,   d0t, d_b0, rb256, nullptr, BB, 256, 128, 0);
        gemm_bf16_wmma<true,  true,  false><<<dim3(4, BB / 128), blk, 0, stream>>>(rb256, d1t, d_b1, rb512, nullptr, BB, 512, 256, 0);
        gemm_bf16_wmma<false, false, true ><<<dim3(1, BB / 128), blk, 0, stream>>>(rb512, d2t, d_b2, nullptr, x_pred + (size_t)(s + 1) * DD, BB, 128, 512, (long long)TT * DD);
    }
}
